// MultiHeadedAttention_2929167696552
// MI455X (gfx1250) — matrix operand order repaired and checked by reading, not compiled, not run on hardware
//
#include <hip/hip_runtime.h>
#include <hip/hip_bf16.h>

// ---------------------------------------------------------------------------
// MultiHeadedAttention on gfx1250 (CDNA5, wave32, WMMA bf16 w/ fp32 accum)
// B=2, S=2048, D=1024, H=16, Dh=64.  SCALE = 1/sqrt(1024) = 0.03125
// ---------------------------------------------------------------------------

typedef __bf16 bf16;
typedef bf16           v16bf __attribute__((ext_vector_type(16)));
typedef float          v8f   __attribute__((ext_vector_type(8)));
typedef unsigned short v16u  __attribute__((ext_vector_type(16)));
typedef unsigned short v8u   __attribute__((ext_vector_type(8)));
typedef unsigned short u16;
typedef unsigned int   u32;

#define WMMA_BF16(A, B, C) \
  __builtin_amdgcn_wmma_f32_16x16x32_bf16(false, (A), false, (B), (short)0, (C), false, false)

static constexpr int   Bc    = 2;
static constexpr int   Sc    = 2048;
static constexpr int   Dc    = 1024;
static constexpr int   Hc    = 16;
static constexpr int   Dh    = 64;
static constexpr int   NQKV  = 3 * Hc * Dh;  // 3072
static constexpr int   Mrows = Bc * Sc;      // 4096
static constexpr float SCALE = 0.03125f;     // 1/sqrt(1024)

__device__ __forceinline__ u16 f2bf(float f) {
  return __builtin_bit_cast(u16, static_cast<bf16>(f));  // hw cvt, RNE
}
__device__ __forceinline__ u32 pack2(float lo, float hi) {
  return (u32)f2bf(lo) | ((u32)f2bf(hi) << 16);
}

__device__ __forceinline__ v8f vzero() {
  v8f z;
#pragma unroll
  for (int i = 0; i < 8; ++i) z[i] = 0.0f;
  return z;
}

// 16 contiguous bf16 (32B, 32B-aligned) -> A/B fragment
__device__ __forceinline__ v16bf fragc(const u16* p) {
  return __builtin_bit_cast(v16bf, *(const v16u*)p);
}

// two 8-half chunks (16B each, 16B-aligned) -> fragment
__device__ __forceinline__ v16bf frag2(const u16* p0, const u16* p1) {
  v8u a = *(const v8u*)p0;
  v8u b = *(const v8u*)p1;
  v16u u;
#pragma unroll
  for (int i = 0; i < 8; ++i) { u[i] = a[i]; u[i + 8] = b[i]; }
  return __builtin_bit_cast(v16bf, u);
}

// ---------------------------------------------------------------------------
// Kernel 1: qkv = x @ W_qkv + b_qkv, scattered into bf16 Q[bh,s,d], K[bh,s,d],
//           and transposed V[bh,d,s].  Tile: 64(M) x 64(N), 4 waves / block.
// ---------------------------------------------------------------------------
__global__ __launch_bounds__(128) void qkv_gemm_kernel(
    const float* __restrict__ x, const float* __restrict__ Wqkv,
    const float* __restrict__ bqkv, u16* __restrict__ qb, u16* __restrict__ kbuf,
    u16* __restrict__ vbuf) {
  __shared__ __align__(32) u16 lds_a[64 * 32];  // [m][k] bf16
  __shared__ __align__(32) u16 lds_b[64 * 32];  // [n][k] bf16 (transposed)

  const int t = threadIdx.x;
  const int lane = t & 31, wave = t >> 5;
  const int l15 = lane & 15;
  const bool hi = lane >= 16;
  const int kb2 = hi ? 8 : 0;    // unused (A-frag also uses kc)
  const int kc  = hi ? 16 : 0;   // B-frag K sub-base
  const int m0 = blockIdx.x * 64;
  const int n0 = blockIdx.y * 64;

  v8f acc[4];
#pragma unroll
  for (int j = 0; j < 4; ++j) acc[j] = vzero();

  for (int k0 = 0; k0 < Dc; k0 += 32) {
    // stage A tile: x[m0..m0+63][k0..k0+31] -> bf16, packed b32 stores
#pragma unroll
    for (int i = 0; i < 8; ++i) {
      int p = t + i * 128;          // pair index: 64 rows x 16 k-pairs
      int r = p >> 4, c2 = (p & 15) * 2;
      const float* xp = &x[(size_t)(m0 + r) * Dc + k0 + c2];
      *(u32*)&lds_a[r * 32 + c2] = pack2(xp[0], xp[1]);
    }
    // stage B tile transposed: W[k0+kk][n0+n] -> lds_b[n][kk]
    // coalesced along n; each thread packs two adjacent k rows
#pragma unroll
    for (int i = 0; i < 8; ++i) {
      int p = t + i * 128;          // pair index: 16 k-pairs x 64 n
      int n = p & 63, kk2 = (p >> 6) * 2;
      const float* wp = &Wqkv[(size_t)(k0 + kk2) * NQKV + n0 + n];
      *(u32*)&lds_b[n * 32 + kk2] = pack2(wp[0], wp[NQKV]);
    }
    __syncthreads();

    const int row = wave * 16 + l15;
    v16bf afrag = fragc(&lds_a[row * 32 + kc]);
#pragma unroll
    for (int nt = 0; nt < 4; ++nt) {
      v16bf bfrag = fragc(&lds_b[(nt * 16 + l15) * 32 + kc]);
      acc[nt] = WMMA_BF16(afrag, bfrag, acc[nt]);
    }
    __syncthreads();
  }

  // epilogue: bias + scatter into Q / K / V(transposed) bf16 buffers
  const int half = hi ? 8 : 0;
#pragma unroll
  for (int nt = 0; nt < 4; ++nt) {
    int n = n0 + nt * 16 + l15;
    int h = n / (3 * Dh);
    int rem = n % (3 * Dh);
    int which = rem / Dh;
    int d = rem % Dh;
    float bias = bqkv[n];
#pragma unroll
    for (int r = 0; r < 8; ++r) {
      int m = m0 + wave * 16 + half + r;
      int b = m >> 11, s = m & (Sc - 1);
      int bh = b * Hc + h;
      u16 val = f2bf(acc[nt][r] + bias);
      if (which == 0)      qb  [((size_t)bh * Sc + s) * Dh + d] = val;
      else if (which == 1) kbuf[((size_t)bh * Sc + s) * Dh + d] = val;
      else                 vbuf[((size_t)bh * Dh + d) * Sc + s] = val;  // V^T
    }
  }
}

// ---------------------------------------------------------------------------
// Kernel 2: flash attention.  grid = (B*H, S/64); 4 waves, each owns a 16-row
// Q tile; stream K/V in 32-column steps with online softmax.
// ---------------------------------------------------------------------------
__global__ __launch_bounds__(128) void attn_kernel(
    const u16* __restrict__ qb, const u16* __restrict__ kbuf,
    const u16* __restrict__ vbuf, const int* __restrict__ mask,
    u16* __restrict__ ctx) {
  __shared__ __align__(32) u16 lds_p[4 * 16 * 32];  // per-wave P tile (bf16)

  const int t = threadIdx.x;
  const int lane = t & 31, wave = t >> 5;
  const int l15 = lane & 15;
  const bool hi = lane >= 16;
  const int kb2 = hi ? 8 : 0;
  const int half = hi ? 8 : 0;

  const int bh = blockIdx.x;   // b*H + h
  const int b = bh >> 4;
  const int h = bh & 15;
  const int q0 = blockIdx.y * 64 + wave * 16;

  // Q A-fragments for d = 0..31 and d = 32..63 (loaded once)
  const u16* qrow = qb + ((size_t)bh * Sc + q0 + l15) * Dh;
  v16bf qf0 = fragc(qrow + (hi ? 16 : 0));
  v16bf qf1 = fragc(qrow + 32 + (hi ? 16 : 0));

  v8f acc[4];
#pragma unroll
  for (int j = 0; j < 4; ++j) acc[j] = vzero();
  float mrow[8], lrow[8];
#pragma unroll
  for (int r = 0; r < 8; ++r) { mrow[r] = -3.0e38f; lrow[r] = 0.0f; }

  const int* mbase = mask + (size_t)b * Sc * Sc;
  u16* pw = &lds_p[wave * 16 * 32];

  for (int kt = 0; kt < Sc; kt += 32) {
    // prefetch next K tile (emits global_prefetch_b8)
    if (kt + 32 < Sc)
      __builtin_prefetch(kbuf + ((size_t)bh * Sc + kt + 32 + l15) * Dh, 0, 3);

    // scores: S = Q @ K^T for 32 columns (two 16-col subtiles)
    v8f sc0 = vzero(), sc1 = vzero();
    {
      const u16* kr = kbuf + ((size_t)bh * Sc + kt + l15) * Dh + (hi ? 16 : 0);
      sc0 = WMMA_BF16(qf0, fragc(kr), sc0);
      sc0 = WMMA_BF16(qf1, fragc(kr + 32), sc0);
      const u16* kr2 = kr + (size_t)16 * Dh;
      sc1 = WMMA_BF16(qf0, fragc(kr2), sc1);
      sc1 = WMMA_BF16(qf1, fragc(kr2 + 32), sc1);
    }

    // online softmax update (rows live per (lane-half, vgpr r))
#pragma unroll
    for (int r = 0; r < 8; ++r) {
      int qg = q0 + half + r;
      float v0 = sc0[r] * SCALE;
      float v1 = sc1[r] * SCALE;
      if (mbase[(size_t)qg * Sc + kt + l15] == 0)      v0 = -1.0e10f;
      if (mbase[(size_t)qg * Sc + kt + 16 + l15] == 0) v1 = -1.0e10f;
      float mx = fmaxf(v0, v1);
#pragma unroll
      for (int off = 1; off < 16; off <<= 1) mx = fmaxf(mx, __shfl_xor(mx, off, 32));
      float mnew = fmaxf(mrow[r], mx);
      float alpha = __expf(mrow[r] - mnew);
      float p0 = __expf(v0 - mnew);
      float p1 = __expf(v1 - mnew);
      float ps = p0 + p1;
#pragma unroll
      for (int off = 1; off < 16; off <<= 1) ps += __shfl_xor(ps, off, 32);
      lrow[r] = lrow[r] * alpha + ps;
      mrow[r] = mnew;
#pragma unroll
      for (int j = 0; j < 4; ++j) acc[j][r] *= alpha;
      pw[(half + r) * 32 + l15] = f2bf(p0);
      pw[(half + r) * 32 + 16 + l15] = f2bf(p1);
    }
    asm volatile("s_wait_dscnt 0" ::: "memory");  // order LDS write -> read (same wave)

    // P (16x32, bf16) as A-fragment; V^T gives contiguous B-fragments
    v16bf pf = fragc(&pw[l15 * 32 + (hi ? 16 : 0)]);
#pragma unroll
    for (int j = 0; j < 4; ++j) {
      const u16* vr = vbuf + ((size_t)bh * Dh + j * 16 + l15) * Sc + kt + (hi ? 16 : 0);
      acc[j] = WMMA_BF16(pf, fragc(vr), acc[j]);
    }
  }

  // epilogue: ctx[b, s, h*64 + d] = acc / l  (bf16)
#pragma unroll
  for (int j = 0; j < 4; ++j) {
    int d = h * Dh + j * 16 + l15;
#pragma unroll
    for (int r = 0; r < 8; ++r) {
      int s = q0 + half + r;
      ctx[((size_t)b * Sc + s) * (Hc * Dh) + d] = f2bf(acc[j][r] / lrow[r]);
    }
  }
}

// ---------------------------------------------------------------------------
// Kernel 3: out = ctx @ W_o + b_o.  M=4096, K=1024, N=64.  64 blocks.
// ---------------------------------------------------------------------------
__global__ __launch_bounds__(128) void out_gemm_kernel(
    const u16* __restrict__ ctx, const float* __restrict__ Wo,
    const float* __restrict__ bo, float* __restrict__ out) {
  __shared__ __align__(32) u16 lds_b[64 * 32];  // W_o tile transposed [n][k]

  const int t = threadIdx.x;
  const int lane = t & 31, wave = t >> 5;
  const int l15 = lane & 15;
  const bool hi = lane >= 16;
  const int kb2 = hi ? 8 : 0;
  const int kc = hi ? 16 : 0;
  const int m0 = blockIdx.x * 64;

  v8f acc[4];
#pragma unroll
  for (int j = 0; j < 4; ++j) acc[j] = vzero();

  for (int k0 = 0; k0 < Hc * Dh; k0 += 32) {
    // stage W_o tile transposed, packed b32 stores
#pragma unroll
    for (int i = 0; i < 8; ++i) {
      int p = t + i * 128;
      int n = p & 63, kk2 = (p >> 6) * 2;
      const float* wp = &Wo[(size_t)(k0 + kk2) * Dh + n];
      *(u32*)&lds_b[n * 32 + kk2] = pack2(wp[0], wp[Dh]);
    }
    __syncthreads();

    const u16* arow = ctx + (size_t)(m0 + wave * 16 + l15) * (Hc * Dh) + k0;
    v16bf afrag = fragc(arow + kc);
#pragma unroll
    for (int nt = 0; nt < 4; ++nt) {
      v16bf bfrag = fragc(&lds_b[(nt * 16 + l15) * 32 + kc]);
      acc[nt] = WMMA_BF16(afrag, bfrag, acc[nt]);
    }
    __syncthreads();
  }

  const int half = hi ? 8 : 0;
#pragma unroll
  for (int nt = 0; nt < 4; ++nt) {
    int n = nt * 16 + l15;
    float bias = bo[n];
#pragma unroll
    for (int r = 0; r < 8; ++r) {
      int m = m0 + wave * 16 + half + r;
      out[(size_t)m * Dh + n] = acc[nt][r] + bias;
    }
  }
}

// ---------------------------------------------------------------------------
extern "C" void kernel_launch(void* const* d_in, const int* in_sizes, int n_in,
                              void* d_out, int out_size, void* d_ws, size_t ws_size,
                              hipStream_t stream) {
  (void)in_sizes; (void)n_in; (void)out_size; (void)ws_size;
  const float* x    = (const float*)d_in[0];
  const float* Wqkv = (const float*)d_in[1];
  const float* bqkv = (const float*)d_in[2];
  const float* Wo   = (const float*)d_in[3];
  const float* bo   = (const float*)d_in[4];
  const int*   mask = (const int*)d_in[5];
  float* out = (float*)d_out;

  const size_t SZ = (size_t)Bc * Hc * Sc * Dh;  // 4,194,304 bf16 elements (8 MB)
  u16* qb   = (u16*)d_ws;
  u16* kbuf = qb + SZ;
  u16* vbuf = kbuf + SZ;
  u16* ctx  = vbuf + SZ;  // [B, S, H*Dh] bf16

  qkv_gemm_kernel<<<dim3(Mrows / 64, NQKV / 64), 128, 0, stream>>>(
      x, Wqkv, bqkv, qb, kbuf, vbuf);
  attn_kernel<<<dim3(Bc * Hc, Sc / 64), 128, 0, stream>>>(
      qb, kbuf, vbuf, mask, ctx);
  out_gemm_kernel<<<Mrows / 64, 128, 0, stream>>>(ctx, Wo, bo, out);
}
